// Transformer_77816217468992
// MI455X (gfx1250) — compile-verified, matrix operand order checked by reading, not run on hardware
//
#include <hip/hip_runtime.h>
#include <hip/hip_bf16.h>
#include <math.h>

// ---------------------------------------------------------------------------
// Types for CDNA5 WMMA (wave32): v_wmma_f32_16x16x32_bf16
// ---------------------------------------------------------------------------
typedef __attribute__((ext_vector_type(16))) __bf16        v16bf;
typedef __attribute__((ext_vector_type(2)))  __bf16        v2bf;
typedef __attribute__((ext_vector_type(8)))  float         v8f;
typedef __attribute__((ext_vector_type(4)))  float         f32x4;
typedef __attribute__((ext_vector_type(4)))  unsigned int  u32x4;
typedef __attribute__((ext_vector_type(2)))  unsigned int  u32x2;

union FragBF { u32x4 q[2]; v16bf v; };

// Problem constants (from reference)
constexpr int B_  = 8,  S_ = 256, T_ = 256;
constexpr int D_  = 1024, H_ = 16, F_ = 4096, L_ = 6;
constexpr int VT_ = 32000, DH_ = 64;

// f32 -> bf16 via native converts (lowers to v_cvt_pk_bf16_f32)
__device__ __forceinline__ unsigned int pk2bf(float a, float b) {
  v2bf t; t.x = (__bf16)a; t.y = (__bf16)b;
  return __builtin_bit_cast(unsigned int, t);
}
__device__ __forceinline__ unsigned short bf1(float a) {
  return __builtin_bit_cast(unsigned short, (__bf16)a);
}

// ---------------------------------------------------------------------------
// General WMMA GEMM:  C[m,n] = alpha * sum_k A[m,k]*B[k,n] (+bias[n]) (relu?)
//   f32 operands converted to bf16 while staging into LDS; WMMA bf16/f32-acc.
//   Batched over blockIdx.z (b,h): offset = zb*s?b + zh*s?h.
//   TRANSB (compile-time): B accessed as B[n*ldb + k].
// Block tile 128 x BN_ (BN_=128 or 64), K-step 32, 256 threads = 8 waves.
// Waves: 4 along M x 2 along N; each wave = 32 x (BN_/2) patch
//   -> 2 x (BN_/32) wmma tiles = 8 (BN_=128) or 4 (BN_=64) WMMAs per K-step.
// Double-buffered LDS + register pipeline: one barrier per K-step.
// ---------------------------------------------------------------------------
constexpr int BM = 128, BK = 32;
constexpr int PITCH = 56;   // shorts; 112B row = 7*16B: aligned, conflict-free

template <int BN_, int TRANSB>
__global__ __launch_bounds__(256) void gemm_wmma_k(
    const float* __restrict__ A, long long lda, long long sAb, long long sAh,
    const float* __restrict__ Bm, long long ldb, long long sBb, long long sBh,
    const float* __restrict__ bias,
    float* __restrict__ C, long long ldc, long long sCb, long long sCh,
    int K, int Hn, float alpha, int relu)
{
  constexpr int TN  = BN_ / 32;              // wmma tiles per wave along N
  constexpr int NB4 = (BN_ * BK) / 4 / 256;  // B float4s per thread
  __shared__ __align__(16) unsigned short As[2][BM * PITCH];
  __shared__ __align__(16) unsigned short Bs[2][BN_ * PITCH];

  const int tid = threadIdx.x;
  const int zb  = blockIdx.z / Hn;
  const int zh  = blockIdx.z % Hn;
  const int m0  = blockIdx.y * BM;
  const int n0  = blockIdx.x * BN_;

  const float* Ab = A  + (long long)zb * sAb + (long long)zh * sAh + (long long)m0 * lda;
  const float* Bb = Bm + (long long)zb * sBb + (long long)zh * sBh;
  float*       Cb = C  + (long long)zb * sCb + (long long)zh * sCh;

  const int wave = tid >> 5;
  const int lane = tid & 31;
  const int wm   = wave & 3;    // 4 waves along M (32 rows each)
  const int wn   = wave >> 2;   // 2 waves along N (BN_/2 cols each)
  const int half = lane >> 4;
  const int l16  = lane & 15;

  // ---- precomputed staging addresses ----
  const float* aP[4];  int aL[4];
  #pragma unroll
  for (int j = 0; j < 4; ++j) {                 // A: 128x32 = 1024 float4s
    int f = tid + j * 256;
    int m = f >> 3, kq = (f & 7) * 4;
    aP[j] = Ab + (long long)m * lda + kq;
    aL[j] = m * PITCH + kq;
  }
  const float* bP[NB4];  int bN[NB4], bK[NB4];
  #pragma unroll
  for (int j = 0; j < NB4; ++j) {
    int f = tid + j * 256;
    if (TRANSB) {                               // float4 along k, row n
      int n = f >> 3, kq = (f & 7) * 4;
      bP[j] = Bb + (long long)(n0 + n) * ldb + kq;
      bN[j] = n; bK[j] = kq;
    } else {                                    // float4 along n, row k
      constexpr int PR = BN_ / 4;
      int k = f / PR, nq = (f % PR) * 4;
      bP[j] = Bb + (long long)k * ldb + n0 + nq;
      bN[j] = nq; bK[j] = k;
    }
  }

  v8f acc[2][TN] = {};
  f32x4 aReg[4], bReg[NB4];

  // ---- prologue: load + store tile 0 ----
  #pragma unroll
  for (int j = 0; j < 4; ++j) aReg[j] = *(const f32x4*)(aP[j]);
  #pragma unroll
  for (int j = 0; j < NB4; ++j) bReg[j] = *(const f32x4*)(bP[j]);
  #pragma unroll
  for (int j = 0; j < 4; ++j) {
    u32x2 pk = { pk2bf(aReg[j].x, aReg[j].y), pk2bf(aReg[j].z, aReg[j].w) };
    *(u32x2*)&As[0][aL[j]] = pk;
  }
  #pragma unroll
  for (int j = 0; j < NB4; ++j) {
    if (TRANSB) {
      u32x2 pk = { pk2bf(bReg[j].x, bReg[j].y), pk2bf(bReg[j].z, bReg[j].w) };
      *(u32x2*)&Bs[0][bN[j] * PITCH + bK[j]] = pk;
    } else {
      Bs[0][(bN[j] + 0) * PITCH + bK[j]] = bf1(bReg[j].x);
      Bs[0][(bN[j] + 1) * PITCH + bK[j]] = bf1(bReg[j].y);
      Bs[0][(bN[j] + 2) * PITCH + bK[j]] = bf1(bReg[j].z);
      Bs[0][(bN[j] + 3) * PITCH + bK[j]] = bf1(bReg[j].w);
    }
  }
  __syncthreads();

  int buf = 0;
  for (int k0 = 0; k0 < K; k0 += BK) {
    const bool hasNext = (k0 + BK) < K;
    // ---- issue next tile's global loads (hide latency under WMMA) ----
    if (hasNext) {
      #pragma unroll
      for (int j = 0; j < 4; ++j) aReg[j] = *(const f32x4*)(aP[j] + (k0 + BK));
      #pragma unroll
      for (int j = 0; j < NB4; ++j)
        bReg[j] = TRANSB ? *(const f32x4*)(bP[j] + (k0 + BK))
                         : *(const f32x4*)(bP[j] + (long long)(k0 + BK) * ldb);
    }
    // ---- compute from current buffer ----
    FragBF afr[2], bfr[TN];
    #pragma unroll
    for (int tm = 0; tm < 2; ++tm) {
      int row = wm * 32 + tm * 16 + l16;
      afr[tm].q[0] = *(const u32x4*)&As[buf][row * PITCH + half * 8];
      afr[tm].q[1] = *(const u32x4*)&As[buf][row * PITCH + 16 + half * 8];
    }
    #pragma unroll
    for (int tn = 0; tn < TN; ++tn) {
      int col = wn * (BN_ / 2) + tn * 16 + l16;
      bfr[tn].q[0] = *(const u32x4*)&Bs[buf][col * PITCH + half * 8];
      bfr[tn].q[1] = *(const u32x4*)&Bs[buf][col * PITCH + 16 + half * 8];
    }
    #pragma unroll
    for (int tm = 0; tm < 2; ++tm)
      #pragma unroll
      for (int tn = 0; tn < TN; ++tn)
        acc[tm][tn] = __builtin_amdgcn_wmma_f32_16x16x32_bf16(
            false, afr[tm].v, false, bfr[tn].v, (short)0, acc[tm][tn], false, false);
    // ---- write next tile into the other buffer ----
    if (hasNext) {
      int nb = buf ^ 1;
      #pragma unroll
      for (int j = 0; j < 4; ++j) {
        u32x2 pk = { pk2bf(aReg[j].x, aReg[j].y), pk2bf(aReg[j].z, aReg[j].w) };
        *(u32x2*)&As[nb][aL[j]] = pk;
      }
      #pragma unroll
      for (int j = 0; j < NB4; ++j) {
        if (TRANSB) {
          u32x2 pk = { pk2bf(bReg[j].x, bReg[j].y), pk2bf(bReg[j].z, bReg[j].w) };
          *(u32x2*)&Bs[nb][bN[j] * PITCH + bK[j]] = pk;
        } else {
          Bs[nb][(bN[j] + 0) * PITCH + bK[j]] = bf1(bReg[j].x);
          Bs[nb][(bN[j] + 1) * PITCH + bK[j]] = bf1(bReg[j].y);
          Bs[nb][(bN[j] + 2) * PITCH + bK[j]] = bf1(bReg[j].z);
          Bs[nb][(bN[j] + 3) * PITCH + bK[j]] = bf1(bReg[j].w);
        }
      }
    }
    __syncthreads();
    buf ^= 1;
  }

  // ---- epilogue: D layout m = r + 8*half, n = l16 ----
  #pragma unroll
  for (int tm = 0; tm < 2; ++tm) {
    #pragma unroll
    for (int tn = 0; tn < TN; ++tn) {
      int col = n0 + wn * (BN_ / 2) + tn * 16 + l16;
      float bv = bias ? bias[col] : 0.0f;
      int rbase = m0 + wm * 32 + tm * 16 + 8 * half;
      #pragma unroll
      for (int r = 0; r < 8; ++r) {
        float v = alpha * acc[tm][tn][r] + bv;
        if (relu) v = fmaxf(v, 0.0f);
        Cb[(long long)(rbase + r) * ldc + col] = v;
      }
    }
  }
}

// ---------------------------------------------------------------------------
// Embedding + positional encoding
// ---------------------------------------------------------------------------
__global__ __launch_bounds__(256) void embed_src_k(const int* __restrict__ src,
                                                   const float* __restrict__ emb,
                                                   float* __restrict__ out) {
  int bs = blockIdx.x;              // over B*S
  int s  = bs % S_;
  long long tok = src[bs];
  const float* e = emb + tok * (long long)D_;
  float* o = out + (long long)bs * D_;
  const float c = -9.210340371976184f / 511.0f;   // -ln(10000)/(D/2-1)
  #pragma unroll
  for (int j = 0; j < D_ / 256; ++j) {
    int d  = threadIdx.x + j * 256;
    int dd = (d < 512) ? d : d - 512;
    float ang = (float)s * __expf((float)dd * c);
    float pe  = (d < 512) ? __sinf(ang) : __cosf(ang);
    o[d] = e[d] * 32.0f + pe;
  }
}

__global__ __launch_bounds__(256) void embed_tgt_k(const int* __restrict__ tgt,
                                                   const float* __restrict__ emb,
                                                   float* __restrict__ out) {
  int bt = blockIdx.x;              // over B*T
  int b = bt / T_, t = bt % T_;
  int tok = (t == 0) ? tgt[b * T_] : tgt[b * T_ + t - 1];   // shifted-right
  const float* e = emb + (long long)tok * D_;
  float* o = out + (long long)bt * D_;
  #pragma unroll
  for (int j = 0; j < D_ / 256; ++j) {
    int d = threadIdx.x + j * 256;
    o[d] = e[d] * 32.0f + ((d < 512) ? 0.0f : 1.0f);        // pos_encoding(1)[0]
  }
}

// ---------------------------------------------------------------------------
// out = LayerNorm(x + delta) * g + b          (delta may be null)
// ---------------------------------------------------------------------------
__global__ __launch_bounds__(256) void add_ln_k(const float* __restrict__ x,
                                                const float* __restrict__ dl,
                                                const float* __restrict__ g,
                                                const float* __restrict__ bb,
                                                float* __restrict__ out) {
  __shared__ float red[256];
  long long row = blockIdx.x;
  const float* xr = x + row * D_;
  const float* dr = dl ? dl + row * D_ : nullptr;
  float loc[D_ / 256];
  float sum = 0.0f;
  #pragma unroll
  for (int j = 0; j < D_ / 256; ++j) {
    int d = threadIdx.x + j * 256;
    float v = xr[d] + (dr ? dr[d] : 0.0f);
    loc[j] = v; sum += v;
  }
  red[threadIdx.x] = sum; __syncthreads();
  for (int o = 128; o > 0; o >>= 1) { if (threadIdx.x < o) red[threadIdx.x] += red[threadIdx.x + o]; __syncthreads(); }
  float mean = red[0] / (float)D_;
  __syncthreads();
  float sq = 0.0f;
  #pragma unroll
  for (int j = 0; j < D_ / 256; ++j) { float t = loc[j] - mean; sq += t * t; }
  red[threadIdx.x] = sq; __syncthreads();
  for (int o = 128; o > 0; o >>= 1) { if (threadIdx.x < o) red[threadIdx.x] += red[threadIdx.x + o]; __syncthreads(); }
  float rstd = rsqrtf(red[0] / (float)D_ + 1e-5f);
  float* orow = out + row * D_;
  #pragma unroll
  for (int j = 0; j < D_ / 256; ++j) {
    int d = threadIdx.x + j * 256;
    orow[d] = (loc[j] - mean) * rstd * g[d] + bb[d];
  }
}

// ---------------------------------------------------------------------------
// Masked softmax over keys.  mode 0: mask on key index s (encoder);
// mode 1: mask on query index t (reference's cross-attn broadcast).
// blockIdx.x = (b*H + h)*T + t ; threadIdx.x = s (S_ == 256)
// ---------------------------------------------------------------------------
__global__ __launch_bounds__(256) void softmax_mask_k(const float* __restrict__ sc,
                                                      float* __restrict__ pr,
                                                      const int* __restrict__ src,
                                                      int mode) {
  __shared__ float red[256];
  long long row = blockIdx.x;
  int t = (int)(row % T_);
  int b = (int)(row / ((long long)H_ * T_));
  int sI = threadIdx.x;
  float v = sc[row * S_ + sI];
  int padTok = (mode == 0) ? src[b * S_ + sI] : src[b * S_ + t];
  if (padTok == 0) v = -1.0e9f;
  red[sI] = v; __syncthreads();
  for (int o = 128; o > 0; o >>= 1) { if (sI < o) red[sI] = fmaxf(red[sI], red[sI + o]); __syncthreads(); }
  float mx = red[0]; __syncthreads();
  float e = __expf(v - mx);
  red[sI] = e; __syncthreads();
  for (int o = 128; o > 0; o >>= 1) { if (sI < o) red[sI] += red[sI + o]; __syncthreads(); }
  pr[row * S_ + sI] = e / red[0];
}

// ---------------------------------------------------------------------------
// In-place log_softmax over last dim (VT_), blockIdx.x over B*T rows
// ---------------------------------------------------------------------------
__global__ __launch_bounds__(256) void log_softmax_k(float* __restrict__ x) {
  __shared__ float red[256];
  long long row = blockIdx.x;
  float* xr = x + row * (long long)VT_;
  float mx = -3.4e38f;
  for (int j = threadIdx.x; j < VT_; j += 256) mx = fmaxf(mx, xr[j]);
  red[threadIdx.x] = mx; __syncthreads();
  for (int o = 128; o > 0; o >>= 1) { if (threadIdx.x < o) red[threadIdx.x] = fmaxf(red[threadIdx.x], red[threadIdx.x + o]); __syncthreads(); }
  mx = red[0]; __syncthreads();
  float s = 0.0f;
  for (int j = threadIdx.x; j < VT_; j += 256) s += __expf(xr[j] - mx);
  red[threadIdx.x] = s; __syncthreads();
  for (int o = 128; o > 0; o >>= 1) { if (threadIdx.x < o) red[threadIdx.x] += red[threadIdx.x + o]; __syncthreads(); }
  float lse = mx + logf(red[0]);
  for (int j = threadIdx.x; j < VT_; j += 256) xr[j] -= lse;
}

// ---------------------------------------------------------------------------
// Host orchestration
// ---------------------------------------------------------------------------
static inline void gemm(hipStream_t st,
    const float* A, long long lda, long long sAb, long long sAh,
    const float* Bm, long long ldb, long long sBb, long long sBh,
    const float* bias, float* C, long long ldc, long long sCb, long long sCh,
    int M, int N, int K, int batch, int Hn, float alpha, int transb, int relu) {
  if (N % 128 == 0) {
    dim3 g(N / 128, M / BM, batch);
    if (transb)
      gemm_wmma_k<128, 1><<<g, 256, 0, st>>>(A, lda, sAb, sAh, Bm, ldb, sBb, sBh,
                                             bias, C, ldc, sCb, sCh, K, Hn, alpha, relu);
    else
      gemm_wmma_k<128, 0><<<g, 256, 0, st>>>(A, lda, sAb, sAh, Bm, ldb, sBb, sBh,
                                             bias, C, ldc, sCb, sCh, K, Hn, alpha, relu);
  } else {
    dim3 g(N / 64, M / BM, batch);
    if (transb)
      gemm_wmma_k<64, 1><<<g, 256, 0, st>>>(A, lda, sAb, sAh, Bm, ldb, sBb, sBh,
                                            bias, C, ldc, sCb, sCh, K, Hn, alpha, relu);
    else
      gemm_wmma_k<64, 0><<<g, 256, 0, st>>>(A, lda, sAb, sAh, Bm, ldb, sBb, sBh,
                                            bias, C, ldc, sCb, sCh, K, Hn, alpha, relu);
  }
}

extern "C" void kernel_launch(void* const* d_in, const int* in_sizes, int n_in,
                              void* d_out, int out_size, void* d_ws, size_t ws_size,
                              hipStream_t stream) {
  (void)in_sizes; (void)n_in; (void)out_size; (void)ws_size;

  const int*   src      = (const int*)  d_in[0];
  const int*   tgt      = (const int*)  d_in[1];
  const float* emb_src  = (const float*)d_in[2];
  const float* emb_tgt  = (const float*)d_in[3];
  const float* e_attn_w = (const float*)d_in[4];
  const float* e_attn_b = (const float*)d_in[5];
  const float* e_ff_w1  = (const float*)d_in[6];
  const float* e_ff_b1  = (const float*)d_in[7];
  const float* e_ff_w2  = (const float*)d_in[8];
  const float* e_ff_b2  = (const float*)d_in[9];
  const float* e_ln_s   = (const float*)d_in[10];
  const float* e_ln_b   = (const float*)d_in[11];
  const float* d_sa_w   = (const float*)d_in[12];
  const float* d_sa_b   = (const float*)d_in[13];
  const float* d_ca_w   = (const float*)d_in[14];
  const float* d_ca_b   = (const float*)d_in[15];
  const float* d_ff_w1  = (const float*)d_in[16];
  const float* d_ff_b1  = (const float*)d_in[17];
  const float* d_ff_w2  = (const float*)d_in[18];
  const float* d_ff_b2  = (const float*)d_in[19];
  const float* d_ln_s   = (const float*)d_in[20];
  const float* d_ln_b   = (const float*)d_in[21];
  const float* Wc       = (const float*)d_in[22];
  const float* bc       = (const float*)d_in[23];
  float* out = (float*)d_out;

  // bump allocator over workspace
  char* p = (char*)d_ws;
  auto alloc = [&](size_t bytes) -> float* {
    float* r = (float*)p; p += (bytes + 255) & ~(size_t)255; return r;
  };
  const size_t BSD = (size_t)B_ * S_ * D_ * sizeof(float);
  float* xe  = alloc(BSD);                                   // encoder hidden / output
  float* y   = alloc(BSD);                                   // decoder hidden
  float* qb  = alloc(BSD);
  float* kb  = alloc(BSD);
  float* vb  = alloc(BSD);
  float* ab  = alloc(BSD);                                   // attn head-concat out
  float* tb  = alloc(BSD);                                   // sublayer output
  float* hid = alloc((size_t)B_ * S_ * F_ * sizeof(float));  // FFN hidden
  float* sco = alloc((size_t)B_ * H_ * T_ * S_ * sizeof(float));
  float* prb = alloc((size_t)B_ * H_ * T_ * S_ * sizeof(float));

  const int ROWS = B_ * S_;                 // 2048
  const long long DD  = (long long)D_ * D_;
  const long long HTS = (long long)H_ * T_ * S_;
  const long long TS  = (long long)T_ * S_;
  const long long TD  = (long long)T_ * D_;
  const long long SD  = (long long)S_ * D_;

  // ----- encoder -----
  embed_src_k<<<B_ * S_, 256, 0, stream>>>(src, emb_src, xe);
  for (int l = 0; l < L_; ++l) {
    const float* aw = e_attn_w + (long long)l * 4 * DD;
    const float* av = e_attn_b + (long long)l * 4 * D_;
    gemm(stream, xe, D_,0,0, aw+0*DD, D_,0,0, av+0*D_, qb, D_,0,0, ROWS, D_, D_, 1,1, 1.f, 0,0);
    gemm(stream, xe, D_,0,0, aw+1*DD, D_,0,0, av+1*D_, kb, D_,0,0, ROWS, D_, D_, 1,1, 1.f, 0,0);
    gemm(stream, xe, D_,0,0, aw+2*DD, D_,0,0, av+2*D_, vb, D_,0,0, ROWS, D_, D_, 1,1, 1.f, 0,0);
    // scores = (Q K^T)/sqrt(DH), batched over (b,h)
    gemm(stream, qb, D_, TD, DH_, kb, D_, SD, DH_, nullptr,
         sco, S_, HTS, TS, T_, S_, DH_, B_*H_, H_, 0.125f, 1, 0);
    softmax_mask_k<<<B_ * H_ * T_, 256, 0, stream>>>(sco, prb, src, 0);
    // out = P V
    gemm(stream, prb, S_, HTS, TS, vb, D_, SD, DH_, nullptr,
         ab, D_, TD, DH_, T_, DH_, S_, B_*H_, H_, 1.f, 0, 0);
    gemm(stream, ab, D_,0,0, aw+3*DD, D_,0,0, av+3*D_, tb, D_,0,0, ROWS, D_, D_, 1,1, 1.f, 0,0);
    add_ln_k<<<ROWS, 256, 0, stream>>>(xe, tb, e_ln_s + (long long)(2*l)*D_,   e_ln_b + (long long)(2*l)*D_,   xe);
    gemm(stream, xe, D_,0,0, e_ff_w1 + (long long)l*D_*F_, F_,0,0, e_ff_b1 + (long long)l*F_,
         hid, F_,0,0, ROWS, F_, D_, 1,1, 1.f, 0, 1);
    gemm(stream, hid, F_,0,0, e_ff_w2 + (long long)l*F_*D_, D_,0,0, e_ff_b2 + (long long)l*D_,
         tb, D_,0,0, ROWS, D_, F_, 1,1, 1.f, 0, 0);
    add_ln_k<<<ROWS, 256, 0, stream>>>(xe, tb, e_ln_s + (long long)(2*l+1)*D_, e_ln_b + (long long)(2*l+1)*D_, xe);
  }

  // ----- decoder -----
  embed_tgt_k<<<B_ * T_, 256, 0, stream>>>(tgt, emb_tgt, y);
  for (int l = 0; l < L_; ++l) {
    const float* sw = d_sa_w + (long long)l * 4 * DD;
    const float* sv = d_sa_b + (long long)l * 4 * D_;
    // simplified self-attn: (h @ w2 + b2) @ w3 + b3
    gemm(stream, y,  D_,0,0, sw+2*DD, D_,0,0, sv+2*D_, qb, D_,0,0, ROWS, D_, D_, 1,1, 1.f, 0,0);
    gemm(stream, qb, D_,0,0, sw+3*DD, D_,0,0, sv+3*D_, tb, D_,0,0, ROWS, D_, D_, 1,1, 1.f, 0,0);
    add_ln_k<<<ROWS, 256, 0, stream>>>(y, tb, d_ln_s + (long long)(3*l)*D_,   d_ln_b + (long long)(3*l)*D_,   y);
    // cross attention: Q from decoder, K/V from encoder output
    const float* cw = d_ca_w + (long long)l * 4 * DD;
    const float* cv = d_ca_b + (long long)l * 4 * D_;
    gemm(stream, y,  D_,0,0, cw+0*DD, D_,0,0, cv+0*D_, qb, D_,0,0, ROWS, D_, D_, 1,1, 1.f, 0,0);
    gemm(stream, xe, D_,0,0, cw+1*DD, D_,0,0, cv+1*D_, kb, D_,0,0, ROWS, D_, D_, 1,1, 1.f, 0,0);
    gemm(stream, xe, D_,0,0, cw+2*DD, D_,0,0, cv+2*D_, vb, D_,0,0, ROWS, D_, D_, 1,1, 1.f, 0,0);
    gemm(stream, qb, D_, TD, DH_, kb, D_, SD, DH_, nullptr,
         sco, S_, HTS, TS, T_, S_, DH_, B_*H_, H_, 0.125f, 1, 0);
    softmax_mask_k<<<B_ * H_ * T_, 256, 0, stream>>>(sco, prb, src, 1);
    gemm(stream, prb, S_, HTS, TS, vb, D_, SD, DH_, nullptr,
         ab, D_, TD, DH_, T_, DH_, S_, B_*H_, H_, 1.f, 0, 0);
    gemm(stream, ab, D_,0,0, cw+3*DD, D_,0,0, cv+3*D_, tb, D_,0,0, ROWS, D_, D_, 1,1, 1.f, 0,0);
    add_ln_k<<<ROWS, 256, 0, stream>>>(y, tb, d_ln_s + (long long)(3*l+1)*D_, d_ln_b + (long long)(3*l+1)*D_, y);
    gemm(stream, y, D_,0,0, d_ff_w1 + (long long)l*D_*F_, F_,0,0, d_ff_b1 + (long long)l*F_,
         hid, F_,0,0, ROWS, F_, D_, 1,1, 1.f, 0, 1);
    gemm(stream, hid, F_,0,0, d_ff_w2 + (long long)l*F_*D_, D_,0,0, d_ff_b2 + (long long)l*D_,
         tb, D_,0,0, ROWS, D_, F_, 1,1, 1.f, 0, 0);
    add_ln_k<<<ROWS, 256, 0, stream>>>(y, tb, d_ln_s + (long long)(3*l+2)*D_, d_ln_b + (long long)(3*l+2)*D_, y);
  }

  // ----- classifier + log_softmax -----
  gemm(stream, y, D_,0,0, Wc, VT_,0,0, bc, out, VT_,0,0, ROWS, VT_, D_, 1,1, 1.f, 0, 0);
  log_softmax_k<<<B_ * T_, 256, 0, stream>>>(out);
}
